// FFLayer_86526411145567
// MI455X (gfx1250) — matrix operand order repaired and checked by reading, not compiled, not run on hardware
//
#include <hip/hip_runtime.h>

// ---------------------------------------------------------------------------
// Real-cosine DFT along dim 1 as batched GEMM on CDNA5 WMMA (gfx1250, wave32).
//   out[b,k,c] = sum_n cos(-2*pi*k*n/N) * x[b,n,c]
// B=16, N=2048 (rows k and reduction n), C=256.
// Split-bf16 (hi+lo) x 3 WMMAs per tile for ~fp32 accuracy in f32 accum.
// Global->LDS tiles moved by the Tensor Data Mover (tensor_load_to_lds),
// double-buffered on TENSORcnt.
// ---------------------------------------------------------------------------

typedef __attribute__((ext_vector_type(16))) __bf16 v16bf;
typedef __attribute__((ext_vector_type(8)))  float  v8f;
typedef __attribute__((ext_vector_type(4)))  unsigned int v4u;
typedef __attribute__((ext_vector_type(8)))  int v8i_;
typedef __attribute__((ext_vector_type(4)))  int v4i_;

#define NK 2048   // k rows of cosine matrix; also reduction length (n)
#define CC 256    // channels
#define NB 16     // batch

#define BM 128    // block tile M (k rows)
#define BN 128    // block tile N (channels)
#define BK 32     // K-step (bf16 WMMA K)
#define KT (NK / BK)   // 64 K iterations

// --------------------------- kernel 1: cos split ---------------------------
// cos_mat[k][n] = cos(-2*pi*((k*n) mod 2048)/2048)  -- exact integer reduction
__global__ void cos_split_kernel(__bf16* __restrict__ chi,
                                 __bf16* __restrict__ clo) {
  unsigned idx = blockIdx.x * 256u + threadIdx.x;   // 0 .. NK*NK-1
  unsigned k = idx >> 11;
  unsigned n = idx & 2047u;
  unsigned m = (k * n) & 2047u;
  float ang = (-6.283185307179586f / 2048.0f) * (float)m;
  float c = cosf(ang);
  __bf16 hi = (__bf16)c;
  __bf16 lo = (__bf16)(c - (float)hi);
  chi[idx] = hi;
  clo[idx] = lo;
}

// ------------------ kernel 2: x split + transpose to [b][c][n] -------------
__global__ void x_split_kernel(const float* __restrict__ x,
                               __bf16* __restrict__ xhi,
                               __bf16* __restrict__ xlo) {
  unsigned idx = blockIdx.x * 256u + threadIdx.x;   // 0 .. NB*NK*CC-1
  unsigned c = idx & (CC - 1);          // low 8 bits
  unsigned n = (idx >> 8) & (NK - 1);   // next 11 bits
  unsigned b = idx >> 19;               // /(NK*CC)
  float v = x[idx];
  __bf16 hi = (__bf16)v;
  __bf16 lo = (__bf16)(v - (float)hi);
  size_t t = ((size_t)b * CC + c) * NK + n;
  xhi[t] = hi;
  xlo[t] = lo;
}

// ----------------------- TDM 2-D tile descriptor ---------------------------
// One tensor_load_to_lds moves a [tile_d1 rows x 32 bf16] tile (row stride
// 2048 elements) from global memory into a contiguous LDS region.
__device__ __forceinline__ void tdm_load_tile32(unsigned lds_addr,
                                                const __bf16* gaddr) {
  const unsigned long long ga = (unsigned long long)(uintptr_t)gaddr;

  v4u g0;
  g0.x = 1u;                                   // count=1 valid, no gather
  g0.y = lds_addr;                             // D#.lds_addr (bytes)
  g0.z = (unsigned)(ga & 0xffffffffu);         // global_addr[31:0]
  g0.w = (unsigned)((ga >> 32) & 0x01ffffffu)  // global_addr[56:32]
         | (2u << 30);                         // type=2 ("image")

  const unsigned tensor_d0 = NK;               // line length (elements)
  const unsigned tensor_d1 = 1u << 20;         // plenty of rows (in-bounds use)
  const unsigned tile_d0   = BK;               // 32 contiguous bf16 per row
  const unsigned tile_d1   = BM;               // 128 rows
  const unsigned long long stride0 = NK;       // elements between rows

  v8i_ g1;
  g1[0] = (int)(1u << 16);                               // data_size=2 bytes
  g1[1] = (int)((tensor_d0 & 0xffffu) << 16);            // tensor_dim0[15:0]
  g1[2] = (int)(((tensor_d0 >> 16) & 0xffffu) |
                ((tensor_d1 & 0xffffu) << 16));          // td0[31:16]|td1[15:0]
  g1[3] = (int)(((tensor_d1 >> 16) & 0xffffu) |
                ((tile_d0 & 0xffffu) << 16));            // td1[31:16]|tile_d0
  g1[4] = (int)(tile_d1 & 0xffffu);                      // tile_d1; tile_d2=0
  g1[5] = (int)(unsigned)(stride0 & 0xffffffffu);        // dim0_stride[31:0]
  g1[6] = (int)(unsigned)((stride0 >> 32) & 0xffffu);    // dim0_stride[47:32]
  g1[7] = 0;                                             // dim1_stride unused

  const v4i_ z4 = {0, 0, 0, 0};
#if defined(__clang_major__) && (__clang_major__ >= 23)
  const v8i_ z8 = {0, 0, 0, 0, 0, 0, 0, 0};
  __builtin_amdgcn_tensor_load_to_lds(g0, g1, z4, z4, z8, 0);
#else
  __builtin_amdgcn_tensor_load_to_lds(g0, g1, z4, z4, 0);
#endif
}

// --------------------------- kernel 3: WMMA GEMM ---------------------------
union Frag16 {
  v16bf v;
  uint4 q[2];
};

__global__ __launch_bounds__(256) void dft_wmma_gemm(
    const __bf16* __restrict__ cosh_, const __bf16* __restrict__ cosl_,
    const __bf16* __restrict__ xth,   const __bf16* __restrict__ xtl,
    float* __restrict__ out) {

  __shared__ alignas(16) __bf16 sAh[2][BM * BK];
  __shared__ alignas(16) __bf16 sAl[2][BM * BK];
  __shared__ alignas(16) __bf16 sBh[2][BN * BK];
  __shared__ alignas(16) __bf16 sBl[2][BN * BK];

  const int tid = threadIdx.x;
  const int m0  = blockIdx.x * BM;   // k-row tile
  const int c0  = blockIdx.y * BN;   // channel tile
  const int b   = blockIdx.z;

  const __bf16* gAh = cosh_ + (size_t)m0 * NK;
  const __bf16* gAl = cosl_ + (size_t)m0 * NK;
  const __bf16* gBh = xth + ((size_t)b * CC + c0) * NK;
  const __bf16* gBl = xtl + ((size_t)b * CC + c0) * NK;

  const int lane = tid & 31;
  const int wave = tid >> 5;           // 8 waves (wave32)
  const int wm = (wave & 1) * 64;      // 2 waves along M, 64 rows each
  const int wn = (wave >> 1) * 32;     // 4 waves along N, 32 cols each
  const int lm = lane & 15;
  const int g  = lane >> 4;            // half-wave select

  v8f acc[4][2];
#pragma unroll
  for (int t = 0; t < 4; ++t)
#pragma unroll
    for (int u = 0; u < 2; ++u)
#pragma unroll
      for (int r = 0; r < 8; ++r) acc[t][u][r] = 0.0f;

  // Issue the 4 TDM tile loads for buffer `buf`, K-step `kt` (wave 0 only;
  // TDM ignores EXEC and issues once per executing wave).
  auto issue_tiles = [&](int buf, int kt) {
    const size_t k0 = (size_t)kt * BK;
    tdm_load_tile32((unsigned)(uintptr_t)&sAh[buf][0], gAh + k0);
    tdm_load_tile32((unsigned)(uintptr_t)&sAl[buf][0], gAl + k0);
    tdm_load_tile32((unsigned)(uintptr_t)&sBh[buf][0], gBh + k0);
    tdm_load_tile32((unsigned)(uintptr_t)&sBl[buf][0], gBl + k0);
  };

  if (tid < 32) issue_tiles(0, 0);

  for (int kt = 0; kt < KT; ++kt) {
    const int cur = kt & 1;

    if (tid < 32) {
      if (kt + 1 < KT) {
        issue_tiles(cur ^ 1, kt + 1);          // prefetch next buffer
        __builtin_amdgcn_s_wait_tensorcnt(4);  // current buffer's 4 done
      } else {
        __builtin_amdgcn_s_wait_tensorcnt(0);
      }
    }
    __syncthreads();   // publish current buffer to all waves

    // A fragments: 16x32 bf16, per lane row m=lm (+half-wave), K-runs at
    // g*8..g*8+7 and 16+g*8..16+g*8+7  (ISA 16-bit A layout)
    Frag16 ah[4], al[4];
#pragma unroll
    for (int t = 0; t < 4; ++t) {
      const int m = wm + t * 16 + lm;
      const __bf16* ph = &sAh[cur][m * BK + g * 8];
      ah[t].q[0] = *(const uint4*)(ph);
      ah[t].q[1] = *(const uint4*)(ph + 16);
      const __bf16* pl = &sAl[cur][m * BK + g * 8];
      al[t].q[0] = *(const uint4*)(pl);
      al[t].q[1] = *(const uint4*)(pl + 16);
    }

    // B fragments: 32x16 bf16, per lane column n=lm, K-runs at g*8..g*8+7
    // and 16+g*8..16+g*8+7 (same order as A)
    Frag16 bh[2], bl[2];
#pragma unroll
    for (int u = 0; u < 2; ++u) {
      const int n = wn + u * 16 + lm;
      const __bf16* ph = &sBh[cur][n * BK + g * 8];
      bh[u].q[0] = *(const uint4*)(ph);
      bh[u].q[1] = *(const uint4*)(ph + 16);
      const __bf16* pl = &sBl[cur][n * BK + g * 8];
      bl[u].q[0] = *(const uint4*)(pl);
      bl[u].q[1] = *(const uint4*)(pl + 16);
    }

    // 3-term split product: hi*hi + hi*lo + lo*hi (lo*lo negligible)
#pragma unroll
    for (int t = 0; t < 4; ++t) {
#pragma unroll
      for (int u = 0; u < 2; ++u) {
        acc[t][u] = __builtin_amdgcn_wmma_f32_16x16x32_bf16(
            false, ah[t].v, false, bh[u].v, (short)0, acc[t][u], false, false);
        acc[t][u] = __builtin_amdgcn_wmma_f32_16x16x32_bf16(
            false, ah[t].v, false, bl[u].v, (short)0, acc[t][u], false, false);
        acc[t][u] = __builtin_amdgcn_wmma_f32_16x16x32_bf16(
            false, al[t].v, false, bh[u].v, (short)0, acc[t][u], false, false);
      }
    }
    __syncthreads();   // all reads of `cur` done before TDM overwrites it
  }

  // store: C/D 16x16 f32 layout -> m = r + 8*g, n = lm
  float* gOut = out + ((size_t)b * NK + m0) * CC + c0;
#pragma unroll
  for (int t = 0; t < 4; ++t)
#pragma unroll
    for (int u = 0; u < 2; ++u)
#pragma unroll
      for (int r = 0; r < 8; ++r) {
        const int m = wm + t * 16 + g * 8 + r;
        const int c = wn + u * 16 + lm;
        gOut[(size_t)m * CC + c] = acc[t][u][r];
      }
}

// ------------------------------- launcher ----------------------------------
extern "C" void kernel_launch(void* const* d_in, const int* in_sizes, int n_in,
                              void* d_out, int out_size, void* d_ws, size_t ws_size,
                              hipStream_t stream) {
  const float* x = (const float*)d_in[0];
  float* out = (float*)d_out;

  char* ws = (char*)d_ws;
  const size_t cosElems = (size_t)NK * NK;        // 4.19M
  const size_t xtElems  = (size_t)NB * CC * NK;   // 8.39M
  __bf16* cos_hi = (__bf16*)(ws);
  __bf16* cos_lo = (__bf16*)(ws + cosElems * 2);
  __bf16* xt_hi  = (__bf16*)(ws + cosElems * 4);
  __bf16* xt_lo  = (__bf16*)(ws + cosElems * 4 + xtElems * 2);

  cos_split_kernel<<<(NK * NK) / 256, 256, 0, stream>>>(cos_hi, cos_lo);
  x_split_kernel<<<(NB * NK * CC) / 256, 256, 0, stream>>>(x, xt_hi, xt_lo);

  dim3 grid(NK / BM, CC / BN, NB);   // 16 x 2 x 16 = 512 workgroups
  dft_wmma_gemm<<<grid, 256, 0, stream>>>(cos_hi, cos_lo, xt_hi, xt_lo, out);
}
